// QuantGRULayer_41188736368934
// MI455X (gfx1250) — matrix operand order repaired and checked by reading, not compiled, not run on hardware
//
#include <hip/hip_runtime.h>
#include <hip/hip_bf16.h>

#define T_STEPS 512
#define BATCH   64
#define IN_DIM  1024
#define HID     1024

typedef __attribute__((ext_vector_type(16))) __bf16 v16bf;
typedef __attribute__((ext_vector_type(8)))  float  v8f;

union Frag16 { unsigned u[8]; uint4 q[2]; v16bf v; };
union Acc    { float f[8];    v8f   v; };

__device__ __forceinline__ unsigned short f2bf(float f) {
  unsigned u = __float_as_uint(f);
  unsigned r = u + 0x7fffu + ((u >> 16) & 1u);   // round-to-nearest-even
  return (unsigned short)(r >> 16);
}
__device__ __forceinline__ float clampf(float x) {
  return fminf(1.0f, fmaxf(-1.0f, x));
}
__device__ __forceinline__ float sigmoidf(float x) {
  return 1.0f / (1.0f + __expf(-x));
}

// A fragment (16x32 bf16) from a pre-packed bf16 row (K-contiguous).
// lane<16: M=lane, K={kb..kb+7, kb+16..kb+23}; lane>=16: M=lane-16, K={+8..+15, +24..+31}
__device__ __forceinline__ void loadAbf(Frag16& a, const unsigned short* __restrict__ row,
                                        int kb, int half) {
  a.q[0] = *(const uint4*)(row + kb + (half ? 8 : 0));
  a.q[1] = *(const uint4*)(row + kb + 16 + (half ? 8 : 0));
}

// B fragment (32x16 bf16): B[k][n] = W[n][k], W row-major (H x K) in bf16.
// lane<16: N=n, K={kb..kb+7, kb+16..kb+23} ; lane>=16: N=n, K={+8..+15, +24..+31}
__device__ __forceinline__ void loadBbf(Frag16& b, const unsigned short* __restrict__ Wbf,
                                        int n, int kb, int half) {
  const unsigned short* p = Wbf + (size_t)n * IN_DIM + kb + (half ? 8 : 0);
  b.q[0] = *(const uint4*)(p);
  b.q[1] = *(const uint4*)(p + 16);
}

#define WMMA_BF16(A, B, C) \
  __builtin_amdgcn_wmma_f32_16x16x32_bf16(false, (A), false, (B), (short)0, (C), false, false)

// ---------------- one-time packs -------------------------------------------------
__global__ void pack_weights_kernel(const float* __restrict__ w0, const float* __restrict__ w1,
                                    const float* __restrict__ w2, const float* __restrict__ w3,
                                    const float* __restrict__ w4, const float* __restrict__ w5,
                                    unsigned short* __restrict__ dst) {
  size_t i = (size_t)blockIdx.x * blockDim.x + threadIdx.x;
  const float* src;
  switch (blockIdx.y) {
    case 0: src = w0; break; case 1: src = w1; break; case 2: src = w2; break;
    case 3: src = w3; break; case 4: src = w4; break; default: src = w5; break;
  }
  dst[(size_t)blockIdx.y * ((size_t)HID * IN_DIM) + i] = f2bf(src[i]);
}

// clamp(X) -> bf16, 4 elements per thread
__global__ void pack_x_kernel(const float* __restrict__ X, unsigned short* __restrict__ Xbf) {
  size_t i = ((size_t)blockIdx.x * blockDim.x + threadIdx.x) * 4;
  float4 x = *(const float4*)(X + i);
  ushort4 o;
  o.x = f2bf(clampf(x.x)); o.y = f2bf(clampf(x.y));
  o.z = f2bf(clampf(x.z)); o.w = f2bf(clampf(x.w));
  *(ushort4*)(Xbf + i) = o;
}

__global__ void init_h_kernel(const float* __restrict__ state,
                              float* __restrict__ h0f, unsigned short* __restrict__ h0bf) {
  size_t i = (size_t)blockIdx.x * blockDim.x + threadIdx.x;
  float h = clampf(state[i]);
  h0f[i]  = h;
  h0bf[i] = f2bf(h);
}

__global__ void copy_last_kernel(const float* __restrict__ src, float* __restrict__ dst) {
  size_t i = (size_t)blockIdx.x * blockDim.x + threadIdx.x;
  dst[i] = src[i];
}

// ---------------- phase 1: Gx = clamp(X)bf16 @ W^T + bias -----------------------
// M = T*B = 32768, N = H, K = I.  128 threads = 4 waves; each wave computes
// 4 M-tiles x 1 N-tile x 3 gates (12 accumulators), reusing the 3 B fragments
// across the 4 A fragments -> 12 WMMAs per 14 b128 loads in the K loop.
__global__ __launch_bounds__(128) void gemm_x_kernel(
    const unsigned short* __restrict__ Xbf,
    const unsigned short* __restrict__ Wri, const unsigned short* __restrict__ Wii,
    const unsigned short* __restrict__ Wni,
    const float* __restrict__ br, const float* __restrict__ bi, const float* __restrict__ bn,
    float* __restrict__ Gr, float* __restrict__ Gi, float* __restrict__ Gn) {
#if defined(__HIP_DEVICE_COMPILE__)
  const int lane = threadIdx.x & 31;
  const int wave = threadIdx.x >> 5;
  const int half = lane >> 4;
  const int l16  = lane & 15;
  const int mBase = blockIdx.y * 256 + wave * 64;   // this wave: rows mBase .. mBase+63
  const int nCol  = blockIdx.x * 16 + l16;

  Acc ar[4], ai[4], an[4];
  v8f z = {};
#pragma unroll
  for (int mt = 0; mt < 4; ++mt) { ar[mt].v = z; ai[mt].v = z; an[mt].v = z; }

  const unsigned short* rowPtr = Xbf + (size_t)(mBase + l16) * IN_DIM;
  for (int kb = 0; kb < IN_DIM; kb += 32) {
    Frag16 b0, b1, b2;
    loadBbf(b0, Wri, nCol, kb, half);
    loadBbf(b1, Wii, nCol, kb, half);
    loadBbf(b2, Wni, nCol, kb, half);
#pragma unroll
    for (int mt = 0; mt < 4; ++mt) {
      Frag16 a;
      loadAbf(a, rowPtr + (size_t)mt * 16 * IN_DIM, kb, half);
      ar[mt].v = WMMA_BF16(a.v, b0.v, ar[mt].v);
      ai[mt].v = WMMA_BF16(a.v, b1.v, ai[mt].v);
      an[mt].v = WMMA_BF16(a.v, b2.v, an[mt].v);
    }
  }

  const float vbr = br[nCol], vbi = bi[nCol], vbn = bn[nCol];
#pragma unroll
  for (int mt = 0; mt < 4; ++mt) {
#pragma unroll
    for (int r = 0; r < 8; ++r) {
      int row = mBase + mt * 16 + r + 8 * half;   // C/D layout: reg r = row r (+8 upper half)
      size_t idx = (size_t)row * HID + nCol;
      Gr[idx] = ar[mt].f[r] + vbr;
      Gi[idx] = ai[mt].f[r] + vbi;
      Gn[idx] = an[mt].f[r] + vbn;                // bias_ni folded (inside later clamp)
    }
  }
#endif
}

// ---------------- phase 2: one recurrent step (512 stream-ordered launches) -----
// Grid = H/16, 128 threads = 4 waves; wave w covers batch rows 16w..16w+15.
// A fragments come from the bf16 shadow of h (written by the previous step).
__global__ __launch_bounds__(128) void gru_step_kernel(
    const unsigned short* __restrict__ hbf_in,   // (B, H) bf16
    const float* __restrict__ hprev,             // (B, H) fp32 (for clamp(h) term)
    const unsigned short* __restrict__ Wrh, const unsigned short* __restrict__ Wih,
    const unsigned short* __restrict__ Wnh,
    const float* __restrict__ bnh,
    const float* __restrict__ Gr, const float* __restrict__ Gi, const float* __restrict__ Gn,
    float* __restrict__ hout, unsigned short* __restrict__ hbf_out) {
#if defined(__HIP_DEVICE_COMPILE__)
  const int lane = threadIdx.x & 31;
  const int wave = threadIdx.x >> 5;
  const int half = lane >> 4;
  const int l16  = lane & 15;
  const int mBase = wave * 16;
  const int nCol  = blockIdx.x * 16 + l16;

  Acc ar, ai, an;
  v8f z = {};
  ar.v = z; ai.v = z; an.v = z;

  const unsigned short* rowPtr = hbf_in + (size_t)(mBase + l16) * HID;
  for (int kb = 0; kb < HID; kb += 32) {
    Frag16 a, b0, b1, b2;
    loadAbf(a, rowPtr, kb, half);                // h enters the matmuls unclamped
    loadBbf(b0, Wrh, nCol, kb, half);
    loadBbf(b1, Wih, nCol, kb, half);
    loadBbf(b2, Wnh, nCol, kb, half);
    ar.v = WMMA_BF16(a.v, b0.v, ar.v);
    ai.v = WMMA_BF16(a.v, b1.v, ai.v);
    an.v = WMMA_BF16(a.v, b2.v, an.v);
  }

  const float vbnh = bnh[nCol];
#pragma unroll
  for (int r = 0; r < 8; ++r) {
    int row = mBase + r + 8 * half;
    size_t idx = (size_t)row * HID + nCol;
    float rg = sigmoidf(Gr[idx] + ar.f[r]);
    float ig = sigmoidf(Gi[idx] + ai.f[r]);
    float nh = an.f[r] + vbnh;
    float ng = tanhf(clampf(Gn[idx]) + clampf(rg * nh));
    float h  = hprev[idx];
    float s  = clampf(h) - ng;
    float hy = ng + clampf(ig * s);
    hout[idx]    = hy;
    hbf_out[idx] = f2bf(hy);                     // bf16 shadow for next step's A frags
  }
#endif
}

extern "C" void kernel_launch(void* const* d_in, const int* in_sizes, int n_in,
                              void* d_out, int out_size, void* d_ws, size_t ws_size,
                              hipStream_t stream) {
  const float* X     = (const float*)d_in[0];
  const float* state = (const float*)d_in[1];
  const float* Wri   = (const float*)d_in[2];
  const float* Wii   = (const float*)d_in[3];
  const float* Wni   = (const float*)d_in[4];
  const float* Wrh   = (const float*)d_in[5];
  const float* Wih   = (const float*)d_in[6];
  const float* Wnh   = (const float*)d_in[7];
  const float* br    = (const float*)d_in[8];
  const float* bi    = (const float*)d_in[9];
  const float* bni   = (const float*)d_in[10];
  const float* bnh   = (const float*)d_in[11];

  const size_t wElems = (size_t)HID * IN_DIM;     // 1M per weight matrix
  const size_t BH     = (size_t)BATCH * HID;      // 64K
  const size_t TBH    = (size_t)T_STEPS * BH;     // 32M
  const size_t TBI    = (size_t)T_STEPS * BATCH * IN_DIM;  // 32M

  // ws layout: [Wbf 12MiB][Xbf 64MiB][h0f][hbf0][hbf1][Gr][Gi][Gn]  (~460 MiB)
  char* ws = (char*)d_ws;
  unsigned short* Wbf  = (unsigned short*)ws;
  unsigned short* Xbf  = Wbf + 6 * wElems;
  float*          h0f  = (float*)(Xbf + TBI);
  unsigned short* hbf0 = (unsigned short*)(h0f + BH);
  unsigned short* hbf1 = hbf0 + BH;
  float* Gr = (float*)(hbf1 + BH);
  float* Gi = Gr + TBH;
  float* Gn = Gi + TBH;

  // 1) one-time packs: weights fp32->bf16 (K-contiguous); clamp(X)->bf16
  pack_weights_kernel<<<dim3((unsigned)(wElems / 256), 6), 256, 0, stream>>>(
      Wri, Wii, Wni, Wrh, Wih, Wnh, Wbf);
  pack_x_kernel<<<(unsigned)(TBI / (256 * 4)), 256, 0, stream>>>(X, Xbf);
  init_h_kernel<<<(unsigned)(BH / 256), 256, 0, stream>>>(state, h0f, hbf0);

  // 2) input projections for all T at once (biases folded)
  gemm_x_kernel<<<dim3(HID / 16, (T_STEPS * BATCH) / 256), 128, 0, stream>>>(
      Xbf, Wbf + 0 * wElems, Wbf + 1 * wElems, Wbf + 2 * wElems,
      br, bi, bni, Gr, Gi, Gn);

  // 3) sequential recurrence: fp32 h lives in d_out rows, bf16 shadow ping-pongs
  float* out = (float*)d_out;
  const unsigned short* WrhB = Wbf + 3 * wElems;
  const unsigned short* WihB = Wbf + 4 * wElems;
  const unsigned short* WnhB = Wbf + 5 * wElems;
  unsigned short* hbf[2] = { hbf0, hbf1 };
  for (int t = 0; t < T_STEPS; ++t) {
    const float* hp = (t == 0) ? h0f : (out + (size_t)(t - 1) * BH);
    gru_step_kernel<<<HID / 16, 128, 0, stream>>>(
        hbf[t & 1], hp, WrhB, WihB, WnhB, bnh,
        Gr + (size_t)t * BH, Gi + (size_t)t * BH, Gn + (size_t)t * BH,
        out + (size_t)t * BH, hbf[(t + 1) & 1]);
  }

  // 4) "last" output = outputs[T-1]
  copy_last_kernel<<<(unsigned)(BH / 256), 256, 0, stream>>>(
      out + (size_t)(T_STEPS - 1) * BH, out + TBH);
}
